// WeightBiGNNDecoder_43241730736196
// MI455X (gfx1250) — matrix operand order repaired and checked by reading, not compiled, not run on hardware
//
#include <hip/hip_runtime.h>
#include <hip/hip_bf16.h>

// ---------- types ----------
typedef _Float16     v16h __attribute__((ext_vector_type(16)));
typedef float        v8f  __attribute__((ext_vector_type(8)));
typedef float        f4   __attribute__((ext_vector_type(4)));
typedef unsigned int u4v  __attribute__((ext_vector_type(4)));

union Frag16 { v16h v; u4v q[2]; _Float16 h[16]; };

// ---------- WMMA GEMM:  C[b] = op(A[b] (.* masked W)) @ B[b] ----------
// Tiles: BM=128, BN=128, BK=32. 256 threads = 8 waves (wave32), wave grid 4(M)x2(N);
// each wave owns a 32x64 C tile = 2x4 fragments -> 8 v_wmma_f32_16x16x32_f16 / K-step.
// Requirements (all call sites satisfy): M%128==0, N%128==0, K%32==0, mlim%32==0.
#define BM 128
#define BN 128
#define BK 32
#define LDA_P 40   // padded K stride in halves -> 80B rows, 16B-aligned frag loads
#define LDB_P 40

template<bool TRANSA, bool MASKED>
__launch_bounds__(256)
__global__ void gemm_wmma_f16(const float* __restrict__ A,
                              const float* __restrict__ W,   // mask weights (unbatched), MASKED only
                              const float* __restrict__ B,
                              float* __restrict__ C,
                              int M, int N, int K,
                              int lda, int ldb, int ldc,
                              long long sA, long long sB, long long sC,
                              int mlim)
{
    __shared__ alignas(16) _Float16 As[BM * LDA_P];
    __shared__ alignas(16) _Float16 Bt[BN * LDB_P];

    const int tid = threadIdx.x;
    const int bz  = blockIdx.z;
    A += sA * bz;  B += sB * bz;  C += sC * bz;

    const int mTile = blockIdx.y * BM;
    const int nTile = blockIdx.x * BN;

    const int wave = tid >> 5;
    const int lane = tid & 31;
    const int hlf  = lane >> 4;        // which 16-lane half
    const int l16  = lane & 15;
    const int wm   = (wave >> 1) * 32; // 0,32,64,96
    const int wn   = (wave & 1) * 64;  // 0,64

    v8f acc[2][4] = {};

    for (int k0 = 0; k0 < K; k0 += BK) {
        // ---- A tile -> As[m][k] (f16), fusing adj*(mask? w : 1) ----
        if (!TRANSA) {
            const int row = tid >> 1;            // 0..127 (m)
            const int cs  = (tid & 1) << 4;      // 0/16   (k)
            const float* ap = A + (size_t)(mTile + row) * lda + (k0 + cs);
            const float* wp = MASKED ? (W + (size_t)(mTile + row) * lda + (k0 + cs)) : nullptr;
            // mlim % BK == 0 -> column-in-mask is uniform over the whole K-tile.
            const bool sel = MASKED && ((mTile + row) < mlim) && (k0 < mlim);
            if (k0 + BK < K) __builtin_prefetch(ap + BK, 0, 1);
            Frag16 st;
            #pragma unroll
            for (int q = 0; q < 4; ++q) {
                f4 av = reinterpret_cast<const f4*>(ap)[q];
                f4 wv = {};
                if (MASKED) wv = reinterpret_cast<const f4*>(wp)[q];
                #pragma unroll
                for (int e = 0; e < 4; ++e) {
                    float x = av[e];
                    if (MASKED && sel) x *= wv[e];
                    st.h[q * 4 + e] = (_Float16)x;
                }
            }
            // two packed 16B LDS stores instead of 16 b16 stores
            *reinterpret_cast<u4v*>(&As[row * LDA_P + cs])     = st.q[0];
            *reinterpret_cast<u4v*>(&As[row * LDA_P + cs + 8]) = st.q[1];
        } else {
            const int kr = tid >> 3;             // 0..31  (k)
            const int cs = (tid & 7) << 4;       // 0..112 (m)
            const float* ap = A + (size_t)(k0 + kr) * lda + (mTile + cs);
            const float* wp = MASKED ? (W + (size_t)(k0 + kr) * lda + (mTile + cs)) : nullptr;
            // mlim % 16 == 0 and the 16-wide column span is 16-aligned -> uniform per thread.
            const bool sel = MASKED && ((k0 + kr) < mlim) && ((mTile + cs) < mlim);
            if (k0 + BK < K) __builtin_prefetch(ap + (size_t)BK * lda, 0, 1);
            #pragma unroll
            for (int q = 0; q < 4; ++q) {
                f4 av = reinterpret_cast<const f4*>(ap)[q];
                f4 wv = {};
                if (MASKED) wv = reinterpret_cast<const f4*>(wp)[q];
                #pragma unroll
                for (int e = 0; e < 4; ++e) {
                    float x = av[e];
                    if (MASKED && sel) x *= wv[e];
                    As[(cs + q * 4 + e) * LDA_P + kr] = (_Float16)x;  // transpose into LDS
                }
            }
        }
        // ---- B tile (32 x BN) -> Bt[n][k] (f16, K-contiguous rows) ----
        {
            const int kr  = tid >> 3;            // 0..31  (k)
            const int ncs = (tid & 7) << 4;      // 0..112 (n)
            const float* bp = B + (size_t)(k0 + kr) * ldb + (nTile + ncs);
            if (k0 + BK < K) __builtin_prefetch(bp + (size_t)BK * ldb, 0, 1);
            #pragma unroll
            for (int q = 0; q < 4; ++q) {
                f4 bv = reinterpret_cast<const f4*>(bp)[q];
                #pragma unroll
                for (int e = 0; e < 4; ++e)
                    Bt[(ncs + q * 4 + e) * LDB_P + kr] = (_Float16)bv[e];
            }
        }
        __syncthreads();

        // ---- fragment loads (ISA 16-bit A 16x32 / B 32x16 layouts) ----
        Frag16 af[2], bf[4];
        #pragma unroll
        for (int i = 0; i < 2; ++i) {
            const int r = wm + 16 * i + l16;
            af[i].q[0] = *reinterpret_cast<const u4v*>(&As[r * LDA_P + 8 * hlf]);       // K = 8h+0..7
            af[i].q[1] = *reinterpret_cast<const u4v*>(&As[r * LDA_P + 16 + 8 * hlf]);  // K = 16+8h+0..7
        }
        #pragma unroll
        for (int j = 0; j < 4; ++j) {
            const int r = wn + 16 * j + l16;
            bf[j].q[0] = *reinterpret_cast<const u4v*>(&Bt[r * LDB_P + 8 * hlf]);       // K = 8h+0..7
            bf[j].q[1] = *reinterpret_cast<const u4v*>(&Bt[r * LDB_P + 16 + 8 * hlf]);  // K = 16+8h+0..7
        }
        #pragma unroll
        for (int i = 0; i < 2; ++i)
            #pragma unroll
            for (int j = 0; j < 4; ++j)
                acc[i][j] = __builtin_amdgcn_wmma_f32_16x16x32_f16(
                    false, af[i].v, false, bf[j].v, (short)0, acc[i][j], false, false);

        __syncthreads();
    }

    // ---- store C (16x16 f32 D layout: lane half selects M+8, reg r = M row) ----
    #pragma unroll
    for (int i = 0; i < 2; ++i)
        #pragma unroll
        for (int j = 0; j < 4; ++j) {
            const int mb = mTile + wm + 16 * i + 8 * hlf;
            const int nb = nTile + wn + 16 * j + l16;
            #pragma unroll
            for (int r = 0; r < 8; ++r)
                C[(size_t)(mb + r) * ldc + nb] = acc[i][j][r];
        }
}

// ---------- row-wise L2 normalize + leaky ReLU over 384-wide rows ----------
__launch_bounds__(256)
__global__ void norm_lrelu_kernel(float* __restrict__ h, int rows)
{
    const int gwave = (int)((blockIdx.x * blockDim.x + threadIdx.x) >> 5);
    const int lane  = threadIdx.x & 31;
    if (gwave >= rows) return;
    float* r = h + (size_t)gwave * 384;
    float vbuf[12];
    float ss = 0.f;
    #pragma unroll
    for (int i = 0; i < 12; ++i) { vbuf[i] = r[lane + 32 * i]; ss += vbuf[i] * vbuf[i]; }
    #pragma unroll
    for (int m = 16; m; m >>= 1) ss += __shfl_xor(ss, m, 32);
    const float nrm = sqrtf(ss);
    const float s   = 1.0f / fmaxf(nrm, 1e-12f);
    #pragma unroll
    for (int i = 0; i < 12; ++i) {
        float x = vbuf[i] * s;
        r[lane + 32 * i] = (x >= 0.f) ? x : 0.1f * x;
    }
}

// ---------- decoder: gather a/b embeddings, t = a p1 p2 p1^T, out = <t, b> ----------
__launch_bounds__(256)
__global__ void decoder_kernel(const float* __restrict__ h,
                               const int* __restrict__ drug,
                               const float* __restrict__ p1,   // (384,256)
                               const float* __restrict__ p2,   // (256,256)
                               float* __restrict__ out)
{
    __shared__ float sa[384], sb[384], su[256], sv[256], st[384], red[256];
    const int b   = blockIdx.x;
    const int tid = threadIdx.x;
    const int ia = drug[b * 2 + 0] - 1;
    const int ib = drug[b * 2 + 1] - 1;
    const float* ha = h + ((size_t)b * 2048 + ia) * 384;
    const float* hb = h + ((size_t)b * 2048 + ib) * 384;
    for (int i = tid; i < 384; i += 256) { sa[i] = ha[i]; sb[i] = hb[i]; }
    __syncthreads();
    { float s = 0.f; for (int k = 0; k < 384; ++k) s += sa[k] * p1[k * 256 + tid]; su[tid] = s; }
    __syncthreads();
    { float s = 0.f; for (int k = 0; k < 256; ++k) s += su[k] * p2[k * 256 + tid]; sv[tid] = s; }
    __syncthreads();
    for (int i = tid; i < 384; i += 256) {
        float s = 0.f;
        for (int k = 0; k < 256; ++k) s += sv[k] * p1[i * 256 + k];
        st[i] = s;
    }
    __syncthreads();
    float s = 0.f;
    for (int i = tid; i < 384; i += 256) s += st[i] * sb[i];
    red[tid] = s; __syncthreads();
    for (int off = 128; off; off >>= 1) { if (tid < off) red[tid] += red[tid + off]; __syncthreads(); }
    if (tid == 0) out[b] = red[0];
}

// ---------- host side ----------
extern "C" void kernel_launch(void* const* d_in, const int* in_sizes, int n_in,
                              void* d_out, int out_size, void* d_ws, size_t ws_size,
                              hipStream_t stream)
{
    (void)in_sizes; (void)n_in; (void)out_size; (void)ws_size;

    const int   NODE = 2048, BATCH = 8, MLIM = 2048 - 64;
    const float* x     = (const float*)d_in[0];
    const int*   drug  = (const int*)d_in[1];
    const float* adj   = (const float*)d_in[2];
    const float* upinv = (const float*)d_in[3];
    const float* dninv = (const float*)d_in[4];
    // layers: {up_w, down_w, bias, up_adj_w, down_adj_w} x 3, then p1, p2
    const float* L[3][5];
    for (int l = 0; l < 3; ++l)
        for (int j = 0; j < 5; ++j)
            L[l][j] = (const float*)d_in[5 + l * 5 + j];
    const float* p1 = (const float*)d_in[20];
    const float* p2 = (const float*)d_in[21];
    float* out = (float*)d_out;

    float* ws = (float*)d_ws;
    const size_t NE = (size_t)BATCH * NODE * 384;   // 6.29M floats per buffer
    float* hA = ws;
    float* hB = ws + NE;
    float* tU = ws + 2 * NE;
    float* tD = ws + 3 * NE;
    float* sU = ws + 4 * NE;
    float* sD = ws + 5 * NE;

    const float* Hin = x;
    int Din = 128;
    float* houts[3] = { hA, hB, hA };

    dim3 blk(256);
    for (int l = 0; l < 3; ++l) {
        float* Hout = houts[l];
        const long long sAdj = (long long)NODE * NODE;
        const long long sH   = (long long)NODE * Din;
        dim3 g1(Din / BN, NODE / BM, BATCH);
        // t_up = (adj .* w_up_masked)^T @ Hin ; t_down = (adj .* w_dn_masked) @ Hin
        gemm_wmma_f16<true,  true ><<<g1, blk, 0, stream>>>(adj, L[l][3], Hin, tU,
            NODE, Din, NODE, NODE, Din, Din, sAdj, sH, sH, MLIM);
        gemm_wmma_f16<false, true ><<<g1, blk, 0, stream>>>(adj, L[l][4], Hin, tD,
            NODE, Din, NODE, NODE, Din, Din, sAdj, sH, sH, MLIM);
        // s = inv_deg @ t
        gemm_wmma_f16<false, false><<<g1, blk, 0, stream>>>(upinv, nullptr, tU, sU,
            NODE, Din, NODE, NODE, Din, Din, sAdj, sH, sH, 0);
        gemm_wmma_f16<false, false><<<g1, blk, 0, stream>>>(dninv, nullptr, tD, sD,
            NODE, Din, NODE, NODE, Din, Din, sAdj, sH, sH, 0);
        // projections, batch-flattened M = 8*2048; write strided into 384-wide concat
        dim3 g2(128 / BN, (BATCH * NODE) / BM, 1);
        gemm_wmma_f16<false, false><<<g2, blk, 0, stream>>>(sU,  nullptr, L[l][0], Hout + 0,
            BATCH * NODE, 128, Din, Din, 128, 384, 0, 0, 0, 0);
        gemm_wmma_f16<false, false><<<g2, blk, 0, stream>>>(sD,  nullptr, L[l][1], Hout + 128,
            BATCH * NODE, 128, Din, Din, 128, 384, 0, 0, 0, 0);
        gemm_wmma_f16<false, false><<<g2, blk, 0, stream>>>(Hin, nullptr, L[l][2], Hout + 256,
            BATCH * NODE, 128, Din, Din, 128, 384, 0, 0, 0, 0);
        // L2 normalize rows + leaky relu
        norm_lrelu_kernel<<<(BATCH * NODE) / 8, blk, 0, stream>>>(Hout, BATCH * NODE);
        Hin = Hout;
        Din = 384;
    }

    decoder_kernel<<<BATCH, blk, 0, stream>>>(houts[2], drug, p1, p2, out);
}
